// EquivariantAttention_62964220559519
// MI455X (gfx1250) — compile-verified, matrix operand order checked by reading, not run on hardware
//
#include <hip/hip_runtime.h>
#include <hip/hip_bf16.h>
#include <math.h>

// ---------------- static problem config (matches reference) ----------------
#define NNODES   10000
#define NEDGES   160000
#define MULTV    16
#define NLV      2
#define DIMV     4
#define EDGE_DIM 32
#define HIDDEN   128
#define RANKV    8
#define NHEAD    4
#define NM       (NLV * MULTV)      // 32
#define LRDIM    (NM * RANKV)       // 256

typedef unsigned int u32;
typedef __bf16 bf16_t;
typedef bf16_t v16bf __attribute__((ext_vector_type(16)));
typedef float  v8f   __attribute__((ext_vector_type(8)));
typedef u32    v8u   __attribute__((ext_vector_type(8)));

// ---------------- small helpers ----------------
__device__ __forceinline__ u32 pack_bf16x2(float lo, float hi) {
  unsigned short a = __builtin_bit_cast(unsigned short, (bf16_t)lo);
  unsigned short b = __builtin_bit_cast(unsigned short, (bf16_t)hi);
  return (u32)a | ((u32)b << 16);
}

__device__ __forceinline__ v8f wmma_bf16(v16bf a, v16bf b, v8f c) {
  // D = A(16x32 bf16) x B(32x16 bf16) + C(16x16 f32)
  return __builtin_amdgcn_wmma_f32_16x16x32_bf16(
      /*neg_a=*/false, a, /*neg_b=*/false, b,
      /*c_mod=*/(short)0, c, /*reuse_a=*/false, /*reuse_b=*/false);
}

// A-matrix 16x32 bf16 fragment, from row-major f32 source (ld floats/row).
// Layout (ISA 7.12.2): lane<16 -> M=lane, K pairs {0..7,16..23};
// lane>=16 -> M=lane-16, K pairs {8..15,24..31}; vgpr i holds K=(k_i, k_i+1).
__device__ __forceinline__ v16bf load_a_f32(const float* src, int ld, int lane) {
  int m  = lane & 15;
  int kb = (lane < 16) ? 0 : 8;
  const float* row = src + (size_t)m * ld;
  v8u r;
#pragma unroll
  for (int i = 0; i < 8; ++i) {
    int k = (i < 4) ? (kb + 2 * i) : (kb + 16 + 2 * (i - 4));
    r[i] = pack_bf16x2(row[k], row[k + 1]);
  }
  return __builtin_bit_cast(v16bf, r);
}

// A fragment from row-major bf16 (viewed as u32 pairs, ld2 = HIDDEN/2 per row).
__device__ __forceinline__ v16bf load_a_h(const u32* h2, int lane, int k0) {
  int m  = lane & 15;
  int kb = k0 + ((lane < 16) ? 0 : 8);
  const u32* row = h2 + (size_t)m * (HIDDEN / 2);
  v8u r;
#pragma unroll
  for (int i = 0; i < 8; ++i) {
    int k = (i < 4) ? (kb + 2 * i) : (kb + 16 + 2 * (i - 4));
    r[i] = row[k >> 1];
  }
  return __builtin_bit_cast(v16bf, r);
}

// B fragment from pre-swizzled weights: frag is 32 lanes x 8 dwords,
// lane-major -> two global_load_b128 per lane.
__device__ __forceinline__ v16bf load_b_frag(const u32* frag, int lane) {
  v8u r = *reinterpret_cast<const v8u*>(frag + (size_t)lane * 8);
  return __builtin_bit_cast(v16bf, r);
}

__device__ __forceinline__ float gelu_exact(float x) {
  return 0.5f * x * (1.0f + erff(x * 0.70710678118654752f));
}

// order-preserving float<->uint key for atomicMax-based segment max
__device__ __forceinline__ u32 fkey(float f) {
  u32 i = __float_as_uint(f);
  return (i & 0x80000000u) ? ~i : (i | 0x80000000u);
}
__device__ __forceinline__ float fdecode(u32 k) {
  u32 i = (k & 0x80000000u) ? (k & 0x7fffffffu) : ~k;
  return __uint_as_float(i);
}

// ---------------- kernel 0: pre-swizzle weights into B-fragment layout ------
// B 32x16 bf16 layout mirrors A: lane n<16 -> N=n, K in {0..7,16..23};
// lane>=16 -> N=lane-16, K in {8..15,24..31}; dword i packs (K=k_i, k_i+1).
// Fragment id = nt*ktiles + kt; storage: frag*256 + lane*8 + i dwords.
__global__ void pack_kernel(const float* __restrict__ w, u32* __restrict__ dst,
                            int Ntot, int ktiles, int ntiles) {
  int tid  = blockIdx.x * blockDim.x + threadIdx.x;
  int i    = tid & 7;
  int lane = (tid >> 3) & 31;
  int frag = tid >> 8;
  if (frag >= ktiles * ntiles) return;
  int kt = frag % ktiles;
  int nt = frag / ktiles;
  int n  = nt * 16 + (lane & 15);
  int kb = kt * 32 + ((lane < 16) ? 0 : 8);
  int k  = (i < 4) ? (kb + 2 * i) : (kb + 16 + 2 * (i - 4));
  dst[tid] = pack_bf16x2(w[(size_t)k * Ntot + n], w[(size_t)(k + 1) * Ntot + n]);
}

// ---------------- kernel 1: h = gelu(ef @ W1 + b1), bf16 out ----------------
// block = 4 waves; each wave owns a 16-edge tile. K = EDGE_DIM = 32 exactly.
__global__ void hidden_kernel(const float* __restrict__ ef,
                              const u32* __restrict__ w1p,
                              const float* __restrict__ bias,
                              unsigned short* __restrict__ hout) {
  int wave = threadIdx.x >> 5, lane = threadIdx.x & 31;
  int e0 = (blockIdx.x * 4 + wave) * 16;
  v16bf a = load_a_f32(ef + (size_t)e0 * EDGE_DIM, EDGE_DIM, lane);
  int n = lane & 15, mb = (lane >> 4) * 8;
#pragma unroll
  for (int nt = 0; nt < HIDDEN / 16; ++nt) {
    v16bf b = load_b_frag(w1p + (size_t)nt * 256, lane);
    v8f c = {};
    c = wmma_bf16(a, b, c);
    float bv = bias[nt * 16 + n];
#pragma unroll
    for (int r = 0; r < 8; ++r) {
      float x = gelu_exact(c[r] + bv);
      hout[(size_t)(e0 + mb + r) * HIDDEN + nt * 16 + n] =
          __builtin_bit_cast(unsigned short, (bf16_t)x);
    }
  }
}

// ---------------- kernel 2: fused left/right GEMM + low-rank apply ---------
// Per wave (16 edges): 128 WMMAs; left/right tiles live in per-wave LDS
// (CDNA5 WGP has 320 KB LDS -> 139 KB/block is fine), then per-edge
// out = B2^T * reshape(left @ (right @ tvec)).
__global__ void conv_kernel(const u32* __restrict__ hbuf2,
                            const u32* __restrict__ wlp, const float* __restrict__ bl,
                            const u32* __restrict__ wrp, const float* __restrict__ br,
                            const float* __restrict__ basis1,   // (E,4,2)
                            const float* __restrict__ basis2,   // (E,2,4)
                            const float* __restrict__ f,        // (N,16,4)
                            const int*   __restrict__ src,
                            float* __restrict__ outbuf) {       // (E,16,4)
  extern __shared__ float smem[];
  int wave = threadIdx.x >> 5, lane = threadIdx.x & 31;
  float* lb = smem + (size_t)wave * 8704;   // 16 x 256
  float* rb = lb + 4096;                    // 16 x 256
  float* tv = rb + 4096;                    // 16 x 32
  int e0 = (blockIdx.x * 4 + wave) * 16;
  int n = lane & 15, mb = (lane >> 4) * 8;

  // --- tvec[e][m*2+l] = sum_d f[src[e]][m][d] * b1[e][d][l] ---
  {
    int e = lane & 15;
    int sn = src[e0 + e];
    const float* fr = f + (size_t)sn * 64;
    const float* bb = basis1 + (size_t)(e0 + e) * 8;
#pragma unroll
    for (int mm = 0; mm < 8; ++mm) {
      int m = mb + mm;
      float s0 = 0.f, s1 = 0.f;
#pragma unroll
      for (int d = 0; d < 4; ++d) {
        float fv = fr[m * 4 + d];
        s0 += fv * bb[d * 2 + 0];
        s1 += fv * bb[d * 2 + 1];
      }
      tv[e * 32 + m * 2 + 0] = s0;
      tv[e * 32 + m * 2 + 1] = s1;
    }
  }

  // --- A fragments of h (16 x 128 bf16), 4 k-steps resident in VGPRs ---
  const u32* h2 = hbuf2 + (size_t)e0 * (HIDDEN / 2);
  v16bf ha[4];
#pragma unroll
  for (int kt = 0; kt < 4; ++kt) ha[kt] = load_a_h(h2, lane, kt * 32);

  // --- left/right = h @ Wl/Wr + bias, tile by tile into LDS ---
  for (int nt = 0; nt < LRDIM / 16; ++nt) {
    v8f cl = {}, cr = {};
#pragma unroll
    for (int kt = 0; kt < 4; ++kt) {
      v16bf bL = load_b_frag(wlp + (size_t)(nt * 4 + kt) * 256, lane);
      cl = wmma_bf16(ha[kt], bL, cl);
      v16bf bR = load_b_frag(wrp + (size_t)(nt * 4 + kt) * 256, lane);
      cr = wmma_bf16(ha[kt], bR, cr);
    }
    __builtin_prefetch(wlp + (size_t)(((nt + 1) & 15) * 4) * 256 + lane * 8, 0, 0);
    __builtin_prefetch(wrp + (size_t)(((nt + 1) & 15) * 4) * 256 + lane * 8, 0, 0);
    float bLv = bl[nt * 16 + n], bRv = br[nt * 16 + n];
#pragma unroll
    for (int r = 0; r < 8; ++r) {
      lb[(size_t)(mb + r) * 256 + nt * 16 + n] = cl[r] + bLv;
      rb[(size_t)(mb + r) * 256 + nt * 16 + n] = cr[r] + bRv;
    }
  }
  __syncthreads();

  // --- per-edge low-rank apply + output basis contraction ---
  {
    int e = lane & 15;
    const float* lrow = lb + (size_t)e * 256;
    const float* rrow = rb + (size_t)e * 256;
    const float* tve  = tv + (size_t)e * 32;
    float r8[RANKV];
#pragma unroll
    for (int p = 0; p < RANKV; ++p) {
      float s = 0.f;
#pragma unroll
      for (int c = 0; c < NM; ++c) s += rrow[p * NM + c] * tve[c];
      r8[p] = s;
    }
    int rbase = (lane >> 4) * 16;
    float ov[16];
#pragma unroll
    for (int rr = 0; rr < 16; ++rr) {
      float s = 0.f;
#pragma unroll
      for (int p = 0; p < RANKV; ++p) s += lrow[(rbase + rr) * RANKV + p] * r8[p];
      ov[rr] = s;
    }
    const float* b2e = basis2 + (size_t)(e0 + e) * 8;
    float* op = outbuf + (size_t)(e0 + e) * 64;
#pragma unroll
    for (int mm = 0; mm < 8; ++mm) {
      int m = mb + mm;
      float o0 = ov[2 * mm], o1 = ov[2 * mm + 1];
#pragma unroll
      for (int d = 0; d < 4; ++d)
        op[m * 4 + d] = o0 * b2e[d] + o1 * b2e[4 + d];
    }
  }
}

// ---------------- kernel 3: Q (equivariant linear) + scratch init ----------
__global__ void qinit_kernel(const float* __restrict__ q_w, const float* __restrict__ q_b,
                             const float* __restrict__ f, float* __restrict__ q,
                             float* __restrict__ out, u32* __restrict__ mmax,
                             float* __restrict__ den) {
  int t = blockIdx.x * blockDim.x + threadIdx.x;
  int node = t >> 4, o = t & 15;
  if (node >= NNODES) return;
  const float* fr = f + (size_t)node * 64;
#pragma unroll
  for (int d = 0; d < 4; ++d) {
    int l = (d == 0) ? 0 : 1;             // IDX = [0,1,1,1]
    const float* wr = q_w + (size_t)(l * 16 + o) * 16;
    float s = 0.f;
#pragma unroll
    for (int m = 0; m < 16; ++m) s += wr[m] * fr[m * 4 + d];
    if (d == 0) s += q_b[o];
    q[(size_t)node * 64 + o * 4 + d] = s;
    out[(size_t)node * 64 + o * 4 + d] = 0.f;
  }
  if (o < NHEAD) {
    mmax[node * NHEAD + o] = 0u;          // key of -inf
    den[node * NHEAD + o]  = 0.f;
  }
}

// ---------------- kernel 4: attention scores + segment max -----------------
__global__ void score_kernel(const float* __restrict__ q, const float* __restrict__ k,
                             const int* __restrict__ dst, float* __restrict__ sc,
                             u32* __restrict__ mmax) {
  int e = blockIdx.x * blockDim.x + threadIdx.x;
  if (e >= NEDGES) return;
  int d = dst[e];
  const float* qr = q + (size_t)d * 64;
  const float* kr = k + (size_t)e * 64;
#pragma unroll
  for (int h = 0; h < NHEAD; ++h) {
    float s = 0.f;
#pragma unroll
    for (int j = 0; j < 16; ++j) s += qr[h * 16 + j] * kr[h * 16 + j];
    s *= 0.25f;                            // 1/sqrt(HM*DIM) = 1/4
    sc[(size_t)e * NHEAD + h] = s;
    atomicMax(&mmax[d * NHEAD + h], fkey(s));
  }
}

// ---------------- kernel 5: exp(s - max) + segment sum ---------------------
__global__ void exps_kernel(const int* __restrict__ dst, float* __restrict__ sc,
                            const u32* __restrict__ mmax, float* __restrict__ den) {
  int e = blockIdx.x * blockDim.x + threadIdx.x;
  if (e >= NEDGES) return;
  int d = dst[e];
#pragma unroll
  for (int h = 0; h < NHEAD; ++h) {
    float m  = fdecode(mmax[d * NHEAD + h]);
    float ex = expf(sc[(size_t)e * NHEAD + h] - m);
    sc[(size_t)e * NHEAD + h] = ex;
    atomicAdd(&den[d * NHEAD + h], ex);
  }
}

// ---------------- kernel 6: weighted scatter of V --------------------------
__global__ void scatter_kernel(const int* __restrict__ dst, const float* __restrict__ sc,
                               const float* __restrict__ den, const float* __restrict__ v,
                               float* __restrict__ out) {
  int id = blockIdx.x * blockDim.x + threadIdx.x;
  int e = id >> 2, h = id & 3;
  if (e >= NEDGES) return;
  int d = dst[e];
  float alpha = sc[(size_t)e * NHEAD + h] / fmaxf(den[d * NHEAD + h], 1e-9f);
  const float* vr = v + (size_t)e * 64 + h * 16;
  float* op = out + (size_t)d * 64 + h * 16;
#pragma unroll
  for (int j = 0; j < 16; ++j) atomicAdd(&op[j], alpha * vr[j]);
}

// ---------------- host launch ----------------
extern "C" void kernel_launch(void* const* d_in, const int* in_sizes, int n_in,
                              void* d_out, int out_size, void* d_ws, size_t ws_size,
                              hipStream_t stream) {
  const float* b1   = (const float*)d_in[0];
  const float* b2   = (const float*)d_in[1];
  const float* ef   = (const float*)d_in[2];
  const float* f    = (const float*)d_in[3];
  const int*   src  = (const int*)d_in[4];
  const int*   dst  = (const int*)d_in[5];
  const float* q_w  = (const float*)d_in[6];
  const float* q_b  = (const float*)d_in[7];
  const float* k_w1 = (const float*)d_in[8];
  const float* k_b1 = (const float*)d_in[9];
  const float* k_wl = (const float*)d_in[10];
  const float* k_bl = (const float*)d_in[11];
  const float* k_wr = (const float*)d_in[12];
  const float* k_br = (const float*)d_in[13];
  const float* v_w1 = (const float*)d_in[14];
  const float* v_b1 = (const float*)d_in[15];
  const float* v_wl = (const float*)d_in[16];
  const float* v_bl = (const float*)d_in[17];
  const float* v_wr = (const float*)d_in[18];
  const float* v_br = (const float*)d_in[19];
  float* out = (float*)d_out;

  char* ws = (char*)d_ws;
  size_t off = 0;
  auto take = [&](size_t bytes) -> char* {
    char* p = ws + off;
    off = (off + bytes + 255) & ~(size_t)255;
    return p;
  };
  u32* kw1p = (u32*)take(8  * 256 * 4);
  u32* kwlp = (u32*)take(64 * 256 * 4);
  u32* kwrp = (u32*)take(64 * 256 * 4);
  u32* vw1p = (u32*)take(8  * 256 * 4);
  u32* vwlp = (u32*)take(64 * 256 * 4);
  u32* vwrp = (u32*)take(64 * 256 * 4);
  unsigned short* hk = (unsigned short*)take((size_t)NEDGES * HIDDEN * 2);
  unsigned short* hv = (unsigned short*)take((size_t)NEDGES * HIDDEN * 2);
  float* kbuf = (float*)take((size_t)NEDGES * 64 * 4);
  float* vbuf = (float*)take((size_t)NEDGES * 64 * 4);
  float* qbuf = (float*)take((size_t)NNODES * 64 * 4);
  float* scb  = (float*)take((size_t)NEDGES * NHEAD * 4);
  u32*   mmx  = (u32*)take((size_t)NNODES * NHEAD * 4);
  float* denb = (float*)take((size_t)NNODES * NHEAD * 4);

  // 0) weight pre-swizzle into WMMA B-fragment order (bf16)
  pack_kernel<<<8,  256, 0, stream>>>(k_w1, kw1p, HIDDEN, 1, 8);
  pack_kernel<<<64, 256, 0, stream>>>(k_wl, kwlp, LRDIM, 4, 16);
  pack_kernel<<<64, 256, 0, stream>>>(k_wr, kwrp, LRDIM, 4, 16);
  pack_kernel<<<8,  256, 0, stream>>>(v_w1, vw1p, HIDDEN, 1, 8);
  pack_kernel<<<64, 256, 0, stream>>>(v_wl, vwlp, LRDIM, 4, 16);
  pack_kernel<<<64, 256, 0, stream>>>(v_wr, vwrp, LRDIM, 4, 16);

  // 1) radial hidden layers (WMMA GEMM + exact GELU)
  hidden_kernel<<<NEDGES / 64, 128, 0, stream>>>(ef, kw1p, k_b1, hk);
  hidden_kernel<<<NEDGES / 64, 128, 0, stream>>>(ef, vw1p, v_b1, hv);

  // 2) fused left/right GEMM + low-rank apply + basis -> K, V
  size_t convShm = (size_t)4 * 8704 * sizeof(float); // 139264 B (< 320 KB WGP LDS)
  conv_kernel<<<NEDGES / 64, 128, convShm, stream>>>(
      (const u32*)hk, kwlp, k_bl, kwrp, k_br, b1, b2, f, src, kbuf);
  conv_kernel<<<NEDGES / 64, 128, convShm, stream>>>(
      (const u32*)hv, vwlp, v_bl, vwrp, v_br, b1, b2, f, src, vbuf);

  // 3) Q + scratch init; 4-6) segment softmax attention + scatter
  qinit_kernel<<<NNODES / 8, 128, 0, stream>>>(q_w, q_b, f, qbuf, out, mmx, denb);
  score_kernel<<<NEDGES / 256, 256, 0, stream>>>(qbuf, kbuf, dst, scb, mmx);
  exps_kernel<<<NEDGES / 256, 256, 0, stream>>>(dst, scb, mmx, denb);
  scatter_kernel<<<(NEDGES * 4) / 256, 256, 0, stream>>>(dst, scb, denb, vbuf, out);
  (void)in_sizes; (void)n_in; (void)out_size; (void)ws_size;
}
